// routing_4458176053395
// MI455X (gfx1250) — matrix operand order repaired and checked by reading, not compiled, not run on hardware
//
#include <hip/hip_runtime.h>
#include <hip/hip_bf16.h>

// Capsule routing for MI455X (gfx1250, wave32).
// GEMMs (x@Wcap and the 4096 votes GEMMs) run on v_wmma_f32_16x16x32_bf16
// with fp32 accumulation; 'votes' is materialized once as bf16 (128MB,
// mostly L2-resident on the 192MB L2) and streamed by the routing kernel
// with global_prefetch ahead of the loads.

#define NS    256   // N samples
#define NIN   128   // input capsules
#define DDIM  256   // input feature dim
#define NOUTC 32    // output capsules
#define INH   64    // in_dim
#define OUTH  64    // out_dim
#define EPSF  1e-8f
#define ITERS 3

typedef __bf16 bf16;
typedef __attribute__((ext_vector_type(4)))  __bf16 v4bf;
typedef __attribute__((ext_vector_type(8)))  __bf16 v8bf;
typedef __attribute__((ext_vector_type(16))) __bf16 v16bf;
typedef __attribute__((ext_vector_type(8)))  float  v8f;

union Frag16 { v16bf v; v8bf h[2]; };
union BfU    { uint4 u; v8bf b; };

__device__ __forceinline__ float bf2f(bf16 b) { return (float)b; }
__device__ __forceinline__ bf16  f2bf(float f) { return (bf16)f; }
__device__ __forceinline__ v4bf f2bf4(float4 f) {
  v4bf v; v[0] = (bf16)f.x; v[1] = (bf16)f.y; v[2] = (bf16)f.z; v[3] = (bf16)f.w;
  return v;
}

// ---------------------------------------------------------------------------
// Workspace layout (bytes)
// ---------------------------------------------------------------------------
#define S_OFF      ((size_t)0)                         // s = x@Wcap+B, fp32 [i][n][h]   8 MB
#define U16_OFF    ((size_t)8*1024*1024)               // u bf16 [i][n][h]               4 MB
#define AI_OFF     ((size_t)12*1024*1024)              // ai fp32 [n][i]               128 KB
#define BIJ_OFF    (AI_OFF + (size_t)NS*NIN*4)         // bij fp32 [n][i][o]             4 MB
#define VOTES_OFF  (BIJ_OFF + (size_t)NS*NIN*NOUTC*4)  // votes bf16 [n][i][o][h]      128 MB

// ---------------------------------------------------------------------------
// Kernel 1: s[i][n][h] = sum_d x[n][i][d] * Wcap[i][d][h] + Bcap[i][h]
// WMMA bf16, tile 16x16x32.  Block = 256 thr (8 waves x 16 rows = 128 rows).
// Software-pipelined: next K-chunk's global loads overlap current WMMAs.
// grid = (2 n-blocks, 128 i)
// ---------------------------------------------------------------------------
#define APAD 40   // row stride (elems) for 32-wide K chunk: 80B, 16B aligned, conflict-free
__global__ void k_proj(const float* __restrict__ x, const float* __restrict__ Wcap,
                       const float* __restrict__ Bcap, float* __restrict__ s) {
  __shared__ bf16 As[128 * APAD];   // 128 rows x 32 K (padded)
  __shared__ bf16 Wt[64 * APAD];    // W^T: 64 h x 32 K (padded)

  const int i     = blockIdx.y;
  const int nbase = blockIdx.x * 128;
  const int tid   = threadIdx.x;
  const int wave  = tid >> 5, lane = tid & 31;
  const int half  = lane >> 4, r = lane & 15;
  const int m0    = wave * 16;

  v8f acc[4] = {};
  float4 xa[4];   // staged A chunk (128x32 / 256 thr = 4 float4 each)
  float4 wa[2];   // staged W chunk (32x64  / 256 thr = 2 float4 each)

  auto load_chunk = [&](int k0) {
    #pragma unroll
    for (int rep = 0; rep < 4; ++rep) {
      int f4 = tid + rep * 256;             // 0..1023
      int row = f4 >> 3, c = (f4 & 7) * 4;
      xa[rep] = *reinterpret_cast<const float4*>(
          &x[((size_t)(nbase + row) * NIN + i) * DDIM + k0 + c]);
    }
    #pragma unroll
    for (int rep = 0; rep < 2; ++rep) {
      int f4 = tid + rep * 256;             // 0..511
      int dl = f4 >> 4, h = (f4 & 15) * 4;
      wa[rep] = *reinterpret_cast<const float4*>(
          &Wcap[((size_t)i * DDIM + k0 + dl) * INH + h]);
    }
  };
  auto stage_lds = [&]() {
    #pragma unroll
    for (int rep = 0; rep < 4; ++rep) {
      int f4 = tid + rep * 256;
      int row = f4 >> 3, c = (f4 & 7) * 4;
      *reinterpret_cast<v4bf*>(&As[row * APAD + c]) = f2bf4(xa[rep]);
    }
    #pragma unroll
    for (int rep = 0; rep < 2; ++rep) {
      int f4 = tid + rep * 256;
      int dl = f4 >> 4, h = (f4 & 15) * 4;
      Wt[(h + 0) * APAD + dl] = f2bf(wa[rep].x);   // transpose scatter
      Wt[(h + 1) * APAD + dl] = f2bf(wa[rep].y);
      Wt[(h + 2) * APAD + dl] = f2bf(wa[rep].z);
      Wt[(h + 3) * APAD + dl] = f2bf(wa[rep].w);
    }
  };

  load_chunk(0);
  for (int k0 = 0; k0 < DDIM; k0 += 32) {
    stage_lds();
    __syncthreads();
    if (k0 + 32 < DDIM) load_chunk(k0 + 32);    // in flight during WMMAs

    // A fragment: lane<16 holds row m0+r, K {0..7,16..23}; lane>=16: K {8..15,24..31}
    Frag16 a;
    {
      int base = (m0 + r) * APAD + half * 8;
      a.h[0] = *reinterpret_cast<const v8bf*>(&As[base]);
      a.h[1] = *reinterpret_cast<const v8bf*>(&As[base + 16]);
    }
    #pragma unroll
    for (int t = 0; t < 4; ++t) {
      // B fragment: lane<16 holds col N=16t+r, K {0..7,16..23}; lane>=16: K {8..15,24..31}
      Frag16 b;
      int base = (16 * t + r) * APAD + half * 8;
      b.h[0] = *reinterpret_cast<const v8bf*>(&Wt[base]);
      b.h[1] = *reinterpret_cast<const v8bf*>(&Wt[base + 16]);
      acc[t] = __builtin_amdgcn_wmma_f32_16x16x32_bf16(
          false, a.v, false, b.v, (short)0, acc[t], false, false);
    }
    __syncthreads();
  }

  // epilogue: + bias, store fp32.  C layout: vgpr e -> M = e + 8*half, N = r.
  #pragma unroll
  for (int t = 0; t < 4; ++t) {
    float bias = Bcap[(size_t)i * INH + 16 * t + r];
    #pragma unroll
    for (int e = 0; e < 8; ++e) {
      int row = m0 + half * 8 + e;
      s[((size_t)i * NS + nbase + row) * INH + 16 * t + r] = acc[t][e] + bias;
    }
  }
}

// ---------------------------------------------------------------------------
// Kernel 2: squash rows of s -> u (bf16) and ai = ||u||.  One wave per (i,n).
// ---------------------------------------------------------------------------
__global__ void k_squash(const float* __restrict__ s, bf16* __restrict__ u16,
                         float* __restrict__ ai) {
  const int wave = threadIdx.x >> 5, lane = threadIdx.x & 31;
  const int flat = blockIdx.x * 8 + wave;      // = i*NS + n
  const float2 sv = *reinterpret_cast<const float2*>(s + (size_t)flat * INH + lane * 2);
  float sq = sv.x * sv.x + sv.y * sv.y;
  #pragma unroll
  for (int off = 16; off > 0; off >>= 1) sq += __shfl_xor(sq, off, 32);
  const float n2 = sq, nrm = sqrtf(sq);
  const float sc = n2 / ((n2 + 1.0f) * (nrm + EPSF));   // squash scale
  u16[(size_t)flat * INH + lane * 2 + 0] = f2bf(sv.x * sc);
  u16[(size_t)flat * INH + lane * 2 + 1] = f2bf(sv.y * sc);
  if (lane == 0) {
    int i = flat >> 8, n = flat & (NS - 1);
    ai[(size_t)n * NIN + i] = sc * nrm;                 // ||squash(s)||
  }
}

// ---------------------------------------------------------------------------
// Kernel 3: votes[n][i][o][:] = (u_i[n] @ Wv_io + Bv_io) * mask[n][i], bf16 out.
// WMMA bf16, K=64 (2 k-steps).  grid = (2 n-blocks, 32 o, 128 i), 256 thr.
// ---------------------------------------------------------------------------
#define VPAD 72   // row stride for 64-wide K: 144B, 16B aligned, conflict-free
__global__ void k_votes(const bf16* __restrict__ u16, const float* __restrict__ Wv,
                        const float* __restrict__ Bv, const float* __restrict__ mask,
                        bf16* __restrict__ votes) {
  __shared__ bf16 As[128 * VPAD];   // 128 n-rows x 64 K
  __shared__ bf16 Wt[64 * VPAD];    // W^T: 64 h x 64 K
  __shared__ bf16 Cs[128 * 64];     // output repack for coalesced stores

  const int i = blockIdx.z, o = blockIdx.y;
  const int nbase = blockIdx.x * 128;
  const int tid = threadIdx.x;
  const int wave = tid >> 5, lane = tid & 31;
  const int half = lane >> 4, r = lane & 15;
  const int m0 = wave * 16;

  // batched loads: A (already bf16) as uint4, W as float4
  BfU av[4];
  #pragma unroll
  for (int rep = 0; rep < 4; ++rep) {
    int fv = tid + rep * 256;           // 0..1023 vec8 chunks
    int row = fv >> 3, c8 = fv & 7;
    av[rep].u = *reinterpret_cast<const uint4*>(
        u16 + ((size_t)i * NS + nbase + row) * INH + c8 * 8);
  }
  float4 wv4[4];
  #pragma unroll
  for (int rep = 0; rep < 4; ++rep) {
    int f4 = tid + rep * 256;           // 0..1023
    int d = f4 >> 4, h = (f4 & 15) * 4;
    wv4[rep] = *reinterpret_cast<const float4*>(
        &Wv[(((size_t)i * NOUTC + o) * INH + d) * OUTH + h]);
  }
  #pragma unroll
  for (int rep = 0; rep < 4; ++rep) {
    int fv = tid + rep * 256;
    int row = fv >> 3, c8 = fv & 7;
    *reinterpret_cast<v8bf*>(&As[row * VPAD + c8 * 8]) = av[rep].b;
  }
  #pragma unroll
  for (int rep = 0; rep < 4; ++rep) {
    int f4 = tid + rep * 256;
    int d = f4 >> 4, h = (f4 & 15) * 4;
    Wt[(h + 0) * VPAD + d] = f2bf(wv4[rep].x);   // transpose scatter
    Wt[(h + 1) * VPAD + d] = f2bf(wv4[rep].y);
    Wt[(h + 2) * VPAD + d] = f2bf(wv4[rep].z);
    Wt[(h + 3) * VPAD + d] = f2bf(wv4[rep].w);
  }
  __syncthreads();

  v8f acc[4] = {};
  #pragma unroll
  for (int ks = 0; ks < 64; ks += 32) {
    Frag16 a;
    int abase = (m0 + r) * VPAD + ks + half * 8;
    a.h[0] = *reinterpret_cast<const v8bf*>(&As[abase]);
    a.h[1] = *reinterpret_cast<const v8bf*>(&As[abase + 16]);
    #pragma unroll
    for (int t = 0; t < 4; ++t) {
      Frag16 b;
      int bbase = (16 * t + r) * VPAD + ks + half * 8;
      b.h[0] = *reinterpret_cast<const v8bf*>(&Wt[bbase]);
      b.h[1] = *reinterpret_cast<const v8bf*>(&Wt[bbase + 16]);
      acc[t] = __builtin_amdgcn_wmma_f32_16x16x32_bf16(
          false, a.v, false, b.v, (short)0, acc[t], false, false);
    }
  }

  // epilogue: +bias, *mask, bf16, repack via LDS
  float mk[8];
  #pragma unroll
  for (int e = 0; e < 8; ++e)
    mk[e] = mask[(size_t)(nbase + m0 + half * 8 + e) * NIN + i];
  #pragma unroll
  for (int t = 0; t < 4; ++t) {
    float bias = Bv[((size_t)i * NOUTC + o) * OUTH + 16 * t + r];
    #pragma unroll
    for (int e = 0; e < 8; ++e) {
      int row = m0 + half * 8 + e;
      Cs[row * 64 + 16 * t + r] = f2bf((acc[t][e] + bias) * mk[e]);
    }
  }
  __syncthreads();

  // coalesced b128 stores: each thread moves 64 bytes of one row
  {
    int row = tid >> 1, hh = (tid & 1) * 32;
    const uint4* src = reinterpret_cast<const uint4*>(&Cs[row * 64 + hh]);
    uint4* dst = reinterpret_cast<uint4*>(
        votes + (((size_t)(nbase + row) * NIN + i) * NOUTC + o) * OUTH + hh);
    #pragma unroll
    for (int q = 0; q < 4; ++q) dst[q] = src[q];
  }
}

__global__ void k_zero(float* __restrict__ p, int count) {
  int idx = blockIdx.x * blockDim.x + threadIdx.x;
  if (idx < count) p[idx] = 0.0f;
}

// ---------------------------------------------------------------------------
// Kernel 4: one routing iteration, one block per sample n (256 threads).
//   Phase 1: cij = ai * softmax([0, bij])[1:]  (per i, into LDS)
//   Phase 2: vj = squash(sum_i cij * votes)    (thread = (o, h-octet))
//   Phase 3: bij += votes . vj                 (skipped on last iteration)
// ---------------------------------------------------------------------------
#define PFD 8   // prefetch distance (i iterations ahead) for votes streaming
__global__ void k_route(const bf16* __restrict__ votes, const float* __restrict__ ai,
                        float* __restrict__ bij, float* __restrict__ out, int last) {
  __shared__ float cij[NIN * NOUTC];   // 16 KB
  const int n = blockIdx.x, t = threadIdx.x;

  if (t < NIN) {
    const int i = t;
    const float* bp = bij + ((size_t)n * NIN + i) * NOUTC;
    float4 bl[8];                                    // one pass over bij row
    #pragma unroll
    for (int q = 0; q < 8; ++q)
      bl[q] = *reinterpret_cast<const float4*>(bp + q * 4);
    float m = 0.0f;                                  // leak logit is 0
    #pragma unroll
    for (int q = 0; q < 8; ++q)
      m = fmaxf(m, fmaxf(fmaxf(bl[q].x, bl[q].y), fmaxf(bl[q].z, bl[q].w)));
    float denom = __expf(-m);                        // leak term
    float e[NOUTC];
    #pragma unroll
    for (int q = 0; q < 8; ++q) {
      e[q * 4 + 0] = __expf(bl[q].x - m);
      e[q * 4 + 1] = __expf(bl[q].y - m);
      e[q * 4 + 2] = __expf(bl[q].z - m);
      e[q * 4 + 3] = __expf(bl[q].w - m);
    }
    #pragma unroll
    for (int o = 0; o < NOUTC; ++o) denom += e[o];
    float sc = ai[(size_t)n * NIN + i] / denom;
    #pragma unroll
    for (int o = 0; o < NOUTC; ++o) cij[i * NOUTC + o] = e[o] * sc;
  }
  __syncthreads();

  const int o = t >> 3, j = t & 7, hb = j * 8;
  const bf16* vb = votes + ((size_t)n * NIN * NOUTC + o) * OUTH + hb;

  float acc[8] = {};
  for (int i = 0; i < NIN; ++i) {
    if (i + PFD < NIN)                               // -> global_prefetch
      __builtin_prefetch((const void*)(vb + (size_t)(i + PFD) * NOUTC * OUTH), 0, 3);
    uint4 raw = *reinterpret_cast<const uint4*>(vb + (size_t)i * NOUTC * OUTH);
    const bf16* vv = reinterpret_cast<const bf16*>(&raw);
    float c = cij[i * NOUTC + o];
    #pragma unroll
    for (int k = 0; k < 8; ++k) acc[k] += c * bf2f(vv[k]);
  }

  // squash across the 8 threads sharing capsule o (lanes j..j^4 within wave)
  float s2 = 0.0f;
  #pragma unroll
  for (int k = 0; k < 8; ++k) s2 += acc[k] * acc[k];
  s2 += __shfl_xor(s2, 1, 32);
  s2 += __shfl_xor(s2, 2, 32);
  s2 += __shfl_xor(s2, 4, 32);
  const float nrm = sqrtf(s2);
  const float sc = s2 / ((s2 + 1.0f) * (nrm + EPSF));

  float vj[8];
  #pragma unroll
  for (int k = 0; k < 8; ++k) vj[k] = acc[k] * sc;

  if (last) {
    #pragma unroll
    for (int k = 0; k < 8; ++k)
      out[((size_t)n * NOUTC + o) * OUTH + hb + k] = vj[k];
    return;
  }

  // agreement: bij += <votes, vj>
  for (int i = 0; i < NIN; ++i) {
    if (i + PFD < NIN)
      __builtin_prefetch((const void*)(vb + (size_t)(i + PFD) * NOUTC * OUTH), 0, 3);
    uint4 raw = *reinterpret_cast<const uint4*>(vb + (size_t)i * NOUTC * OUTH);
    const bf16* vv = reinterpret_cast<const bf16*>(&raw);
    float d = 0.0f;
    #pragma unroll
    for (int k = 0; k < 8; ++k) d += bf2f(vv[k]) * vj[k];
    d += __shfl_xor(d, 1, 32);
    d += __shfl_xor(d, 2, 32);
    d += __shfl_xor(d, 4, 32);
    if (j == 0) bij[((size_t)n * NIN + i) * NOUTC + o] += d;
  }
}

// ---------------------------------------------------------------------------
extern "C" void kernel_launch(void* const* d_in, const int* in_sizes, int n_in,
                              void* d_out, int out_size, void* d_ws, size_t ws_size,
                              hipStream_t stream) {
  (void)in_sizes; (void)n_in; (void)out_size; (void)ws_size;
  const float* x     = (const float*)d_in[0];  // [N, NIN, D]
  const float* mask  = (const float*)d_in[1];  // [N, NIN]
  const float* Wcap  = (const float*)d_in[2];  // [NIN, D, INH]
  const float* Bcap  = (const float*)d_in[3];  // [NIN, INH]
  const float* Wv    = (const float*)d_in[4];  // [NIN, NOUTC, INH, OUTH]
  const float* Bv    = (const float*)d_in[5];  // [NIN, NOUTC, 1, OUTH]
  float* out = (float*)d_out;                  // [N, NOUTC, OUTH]

  char* ws = (char*)d_ws;
  float* s     = (float*)(ws + S_OFF);
  bf16*  u16   = (bf16*)(ws + U16_OFF);
  float* ai    = (float*)(ws + AI_OFF);
  float* bij   = (float*)(ws + BIJ_OFF);
  bf16*  votes = (bf16*)(ws + VOTES_OFF);

  k_proj<<<dim3(2, NIN), 256, 0, stream>>>(x, Wcap, Bcap, s);
  k_squash<<<(NS * NIN) / 8, 256, 0, stream>>>(s, u16, ai);
  k_zero<<<(NS * NIN * NOUTC) / 256, 256, 0, stream>>>(bij, NS * NIN * NOUTC);
  k_votes<<<dim3(2, NOUTC, NIN), 256, 0, stream>>>(u16, Wv, Bv, mask, votes);
  for (int it = 0; it < ITERS; ++it)
    k_route<<<NS, 256, 0, stream>>>(votes, ai, bij, out, it == ITERS - 1 ? 1 : 0);
}
